// GINEdgeTunable_6648609374620
// MI455X (gfx1250) — matrix operand order repaired and checked by reading, not compiled, not run on hardware
//
#include <hip/hip_runtime.h>
#include <hip/hip_bf16.h>

// ---- problem constants (match reference) ----
#define NNODES   300000
#define NEDGES   600000
#define HD       128
#define NLAYERS  5
#define NGRAPHS  8192
#define BN_EPS_F 1e-5f

typedef __attribute__((ext_vector_type(16))) __bf16 v16bf;
typedef __attribute__((ext_vector_type(8)))  float  v8f;

static __device__ __forceinline__ __bf16 f2bf(float f) { return (__bf16)f; }

// -------------------- utility kernels --------------------
__global__ void k_zero4(float4* __restrict__ p, long n4) {
  long i = (long)blockIdx.x * blockDim.x + threadIdx.x;
  if (i < n4) p[i] = make_float4(0.f, 0.f, 0.f, 0.f);
}

// h[i,:] = sum_f atom_emb[f, x[i,f], :]   (4 cols per thread)
__global__ void k_atom_encode(const int* __restrict__ x, const float4* __restrict__ aemb4,
                              float4* __restrict__ h4, long total4) {
  long idx = (long)blockIdx.x * blockDim.x + threadIdx.x;
  if (idx >= total4) return;
  int i  = (int)(idx >> 5);       // node
  int c4 = (int)(idx & 31);       // float4 column chunk
  float4 s = make_float4(0.f, 0.f, 0.f, 0.f);
#pragma unroll
  for (int f = 0; f < 9; ++f) {
    int v = x[i * 9 + f];
    float4 t = aemb4[((size_t)f * 128 + v) * 32 + c4];
    s.x += t.x; s.y += t.y; s.z += t.z; s.w += t.w;
  }
  h4[idx] = s;
}

__global__ void k_deg(const int* __restrict__ row, float* __restrict__ deg, int E) {
  int e = blockIdx.x * blockDim.x + threadIdx.x;
  if (e < E) atomicAdd(deg + row[e], 1.0f);
}

__global__ void k_inv_inplace(float* __restrict__ d, int n) {
  int i = blockIdx.x * blockDim.x + threadIdx.x;
  if (i < n) d[i] = 1.0f / (d[i] + 1.0f);
}

// acc[row[e],:] += bond_emb_l[0,ea0,:] + bond_emb_l[1,ea1,:] + bond_emb_l[2,ea2,:]
__global__ void k_bond_scatter(const int* __restrict__ eattr, const int* __restrict__ row,
                               const float* __restrict__ bemb, float* __restrict__ acc,
                               long total) {
  long idx = (long)blockIdx.x * blockDim.x + threadIdx.x;
  if (idx >= total) return;
  int e  = (int)(idx >> 7);
  int hh = (int)(idx & 127);
  int r  = row[e];
  float v = bemb[(0 * 8 + eattr[e * 3 + 0]) * 128 + hh]
          + bemb[(1 * 8 + eattr[e * 3 + 1]) * 128 + hh]
          + bemb[(2 * 8 + eattr[e * 3 + 2]) * 128 + hh];
  atomicAdd(acc + (size_t)r * 128 + hh, v);
}

// h_in = h + acc*inv[node]; then re-zero acc for the next scatter (float4)
__global__ void k_hin(const float4* __restrict__ h, float4* __restrict__ acc,
                      const float* __restrict__ inv, float4* __restrict__ hin, long total4) {
  long idx = (long)blockIdx.x * blockDim.x + threadIdx.x;
  if (idx >= total4) return;
  float iv = inv[idx >> 5];
  float4 a = acc[idx], hv = h[idx];
  hin[idx] = make_float4(hv.x + a.x * iv, hv.y + a.y * iv, hv.z + a.z * iv, hv.w + a.w * iv);
  acc[idx] = make_float4(0.f, 0.f, 0.f, 0.f);
}

// acc[col[e],:] += h_in[row[e],:]
__global__ void k_aggr_scatter(const int* __restrict__ row, const int* __restrict__ col,
                               const float* __restrict__ hin, float* __restrict__ acc,
                               long total) {
  long idx = (long)blockIdx.x * blockDim.x + threadIdx.x;
  if (idx >= total) return;
  int e  = (int)(idx >> 7);
  int hh = (int)(idx & 127);
  float v = hin[(size_t)row[e] * 128 + hh];
  atomicAdd(acc + (size_t)col[e] * 128 + hh, v);
}

// Wt_bf16[n,k] = bf16(W_f32[k,n]); also zero the 256-entry stats buffer
__global__ void k_prep_w(const float* __restrict__ W, __bf16* __restrict__ Wt,
                         float* __restrict__ stats) {
  int idx = blockIdx.x * blockDim.x + threadIdx.x;  // 64 blocks * 256 = 16384
  int k = idx >> 7, n = idx & 127;
  Wt[(size_t)n * 128 + k] = f2bf(W[idx]);
  if (idx < 256) stats[idx] = 0.0f;
}

// -------------------- WMMA GEMM: Y[M,128] = f(X)[M,128] @ W[128,128] --------------------
// IN_MODE: 0 = A from X1 (+bias in epilogue, optional relu)
//          1 = A = (1+eps[lidx])*X1 + X2      (fused GIN combine)
//          2 = A = relu(X1*aux[k] + aux[128+k]) (fused BN+ReLU on input)
// BN'd outputs (DO_STATS) need no bias: BatchNorm cancels it (mean shift,
// variance invariant). 256 threads/8 waves, 128x128 tile; W^T bf16 in LDS;
// 4 chained v_wmma_f32_16x16x32_bf16 per column tile; fused column sum/sumsq
// via ds_add_f32 + one global atomic per column per block.
template <int IN_MODE, bool DO_STATS, bool DO_RELU>
__global__ void k_gemm128(const float* __restrict__ X1, const float* __restrict__ X2,
                          const float* __restrict__ aux, int lidx,
                          const __bf16* __restrict__ Wt, const float* __restrict__ bias,
                          float* __restrict__ Y, float* __restrict__ stats, int M) {
  __shared__ __align__(16) __bf16 sW[128 * 128];   // 32KB: W^T tile
  __shared__ float lsum[128];
  __shared__ float lsq[128];
  int tid = threadIdx.x;

  // cooperative stage of W^T into LDS (16384 bf16 = 2048 x 16B)
  {
    const uint4* src = (const uint4*)Wt;
    uint4* dst = (uint4*)sW;
#pragma unroll
    for (int t = 0; t < 8; ++t) dst[tid + 256 * t] = src[tid + 256 * t];
  }
  if (DO_STATS && tid < 128) { lsum[tid] = 0.0f; lsq[tid] = 0.0f; }
  __syncthreads();

  int wave = tid >> 5;
  int lane = tid & 31;
  int half = lane >> 4;    // lane group 0/1
  int lm   = lane & 15;    // row (A) / col (B,C) within 16x16 tile
  int r0   = blockIdx.x * 128 + wave * 16;
  // wave-uniform full-tile flag, forced into an SGPR -> scalar branches
  int full = __builtin_amdgcn_readfirstlane((int)((r0 + 16) <= M));

  float ecoef = 0.0f;
  if (IN_MODE == 1) ecoef = 1.0f + aux[lidx];

  // ---- load A fragments (16 rows x 128 K) as bf16, contiguous k order (k = 16*half + s) ----
  // lanes 0-15:  K = kb*32 + [0..15]
  // lanes 16-31: K = kb*32 + [16..31]
  int  arow  = r0 + lm;
  bool rowok = arow < M;
  int  crow  = rowok ? arow : (M > 0 ? M - 1 : 0);
  const float* rp1 = X1 + (size_t)crow * 128;
  const float* rp2 = (IN_MODE == 1) ? (X2 + (size_t)crow * 128) : nullptr;

  auto ldA = [&](int k) -> float {
    float v;
    if (IN_MODE == 0) {
      v = rp1[k];
    } else if (IN_MODE == 1) {
      v = fmaf(ecoef, rp1[k], rp2[k]);
    } else {
      v = fmaf(rp1[k], aux[k], aux[128 + k]);
      v = fmaxf(v, 0.0f);
    }
    return v;
  };

  v16bf afrag[4];
  if (full) {
#pragma unroll
    for (int kb = 0; kb < 4; ++kb) {
      int b0 = kb * 32 + half * 16;
      v16bf a;
#pragma unroll
      for (int t = 0; t < 8; ++t) {
        a[t]     = f2bf(ldA(b0 + t));
        a[8 + t] = f2bf(ldA(b0 + 8 + t));
      }
      afrag[kb] = a;
    }
  } else {
#pragma unroll
    for (int kb = 0; kb < 4; ++kb) {
      int b0 = kb * 32 + half * 16;
      v16bf a;
#pragma unroll
      for (int t = 0; t < 8; ++t) {
        a[t]     = f2bf(rowok ? ldA(b0 + t) : 0.0f);
        a[8 + t] = f2bf(rowok ? ldA(b0 + 8 + t) : 0.0f);
      }
      afrag[kb] = a;
    }
  }

  // ---- 8 column tiles ----
#pragma unroll 1
  for (int j = 0; j < 8; ++j) {
    int n = j * 16 + lm;
    const __bf16* bp = sW + (size_t)n * 128 + half * 16;
    v16bf b0 = *(const v16bf*)(bp);
    v16bf b1 = *(const v16bf*)(bp + 32);
    v16bf b2 = *(const v16bf*)(bp + 64);
    v16bf b3 = *(const v16bf*)(bp + 96);
    v8f c = {0.0f, 0.0f, 0.0f, 0.0f, 0.0f, 0.0f, 0.0f, 0.0f};
    c = __builtin_amdgcn_wmma_f32_16x16x32_bf16(false, afrag[0], false, b0, (short)0, c, false, false);
    c = __builtin_amdgcn_wmma_f32_16x16x32_bf16(false, afrag[1], false, b1, (short)0, c, false, false);
    c = __builtin_amdgcn_wmma_f32_16x16x32_bf16(false, afrag[2], false, b2, (short)0, c, false, false);
    c = __builtin_amdgcn_wmma_f32_16x16x32_bf16(false, afrag[3], false, b3, (short)0, c, false, false);

    float bnv = 0.0f;
    if (!DO_STATS) bnv = bias[n];      // BN'd GEMMs: bias cancels in BatchNorm
    float s = 0.0f, sq = 0.0f;
    if (full) {
#pragma unroll
      for (int r = 0; r < 8; ++r) {
        int m = r + half * 8;               // lanes 16-31 hold rows 8..15
        float y = c[r];
        if (!DO_STATS) y += bnv;
        if (DO_RELU) y = fmaxf(y, 0.0f);
        Y[(size_t)(r0 + m) * 128 + n] = y;
        if (DO_STATS) { s += y; sq += y * y; }
      }
    } else {
#pragma unroll
      for (int r = 0; r < 8; ++r) {
        int m    = r + half * 8;
        int grow = r0 + m;
        if (grow < M) {
          float y = c[r];
          if (!DO_STATS) y += bnv;
          if (DO_RELU) y = fmaxf(y, 0.0f);
          Y[(size_t)grow * 128 + n] = y;
          if (DO_STATS) { s += y; sq += y * y; }
        }
      }
    }
    if (DO_STATS) {
      atomicAdd(&lsum[n], s);
      atomicAdd(&lsq[n], sq);
    }
  }

  if (DO_STATS) {
    __syncthreads();
    if (tid < 128) {
      atomicAdd(&stats[tid], lsum[tid]);
      atomicAdd(&stats[128 + tid], lsq[tid]);
    }
  }
}

// fold BN params over bias-free activations: scale = g*rsqrt(var+eps),
// shift = b - mean*scale (variance is bias-invariant, mean shift cancels)
__global__ void k_bn_finalize(const float* __restrict__ stats, const float* __restrict__ g,
                              const float* __restrict__ b, float* __restrict__ ab, float invN) {
  int i = threadIdx.x;  // 128
  float mean = stats[i] * invN;
  float var  = stats[128 + i] * invN - mean * mean;
  float a    = g[i] * rsqrtf(var + BN_EPS_F);
  ab[i]       = a;
  ab[128 + i] = b[i] - mean * a;
}

// out = bn(Y) (+relu), float4
__global__ void k_bn_apply(const float4* __restrict__ Y, const float* __restrict__ ab,
                           float4* __restrict__ out, long total4, int relu) {
  long idx = (long)blockIdx.x * blockDim.x + threadIdx.x;
  if (idx >= total4) return;
  int n0 = (int)(idx & 31) * 4;
  float4 y = Y[idx];
  float4 v = make_float4(fmaf(y.x, ab[n0 + 0], ab[128 + n0 + 0]),
                         fmaf(y.y, ab[n0 + 1], ab[128 + n0 + 1]),
                         fmaf(y.z, ab[n0 + 2], ab[128 + n0 + 2]),
                         fmaf(y.w, ab[n0 + 3], ab[128 + n0 + 3]));
  if (relu) {
    v.x = fmaxf(v.x, 0.f); v.y = fmaxf(v.y, 0.f);
    v.z = fmaxf(v.z, 0.f); v.w = fmaxf(v.w, 0.f);
  }
  out[idx] = v;
}

// pooled[batch[i],:] += h[i,:]
__global__ void k_pool(const int* __restrict__ batch, const float* __restrict__ h,
                       float* __restrict__ pooled, long total) {
  long idx = (long)blockIdx.x * blockDim.x + threadIdx.x;
  if (idx >= total) return;
  int i  = (int)(idx >> 7);
  int hh = (int)(idx & 127);
  atomicAdd(pooled + (size_t)batch[i] * 128 + hh, h[idx]);
}

// out[g] = dot(hid[g,:], cw2) + cb2  -- one wave32 per graph
__global__ void k_final_dot(const float* __restrict__ hid, const float* __restrict__ cw2,
                            const float* __restrict__ cb2, float* __restrict__ out, int Gn) {
  int g    = blockIdx.x * 8 + (threadIdx.x >> 5);
  int lane = threadIdx.x & 31;
  if (g >= Gn) return;
  const float* hp = hid + (size_t)g * 128;
  float s = 0.0f;
#pragma unroll
  for (int t = 0; t < 4; ++t) s += hp[lane + 32 * t] * cw2[lane + 32 * t];
#pragma unroll
  for (int off = 16; off > 0; off >>= 1) s += __shfl_down(s, off, 32);
  if (lane == 0) out[g] = s + cb2[0];
}

// -------------------- launch --------------------
extern "C" void kernel_launch(void* const* d_in, const int* in_sizes, int n_in,
                              void* d_out, int out_size, void* d_ws, size_t ws_size,
                              hipStream_t stream) {
  (void)in_sizes; (void)n_in; (void)out_size; (void)ws_size;
  const int  N = NNODES, E = NEDGES, Gn = NGRAPHS;
  const long NH = (long)N * HD;
  const long EH = (long)E * HD;
  const long GH = (long)Gn * HD;

  // inputs (setup_inputs order)
  const int*   x         = (const int*)  d_in[0];
  const int*   erow      = (const int*)  d_in[1];            // edge_index[0,:]
  const int*   ecol      = ((const int*) d_in[1]) + E;       // edge_index[1,:]
  const int*   eattr     = (const int*)  d_in[2];
  const int*   batch     = (const int*)  d_in[3];
  const float* atom_emb  = (const float*)d_in[4];
  const float* bond_emb  = (const float*)d_in[5];
  const float* eps       = (const float*)d_in[6];
  const float* w1        = (const float*)d_in[7];
  const float* bn1_g     = (const float*)d_in[9];
  const float* bn1_b     = (const float*)d_in[10];
  const float* w2        = (const float*)d_in[11];
  const float* bn_g      = (const float*)d_in[13];
  const float* bn_b      = (const float*)d_in[14];
  const float* cw1       = (const float*)d_in[15];
  const float* cb1       = (const float*)d_in[16];
  const float* cw2       = (const float*)d_in[17];
  const float* cb2       = (const float*)d_in[18];
  float* out = (float*)d_out;

  // workspace carve-up (256B aligned)
  char* ws = (char*)d_ws;
  size_t off = 0;
  auto take = [&](size_t bytes) { char* p = ws + off; off = (off + bytes + 255) & ~(size_t)255; return p; };
  float*  b_h    = (float*) take(NH * 4);
  float*  b_acc  = (float*) take(NH * 4);
  float*  b_hin  = (float*) take(NH * 4);
  float*  b_inv  = (float*) take((size_t)N * 4);
  float*  b_pool = (float*) take(GH * 4);
  __bf16* b_wt   = (__bf16*)take((size_t)HD * HD * 2);
  float*  b_st   = (float*) take(256 * 4);
  float*  b_ab1  = (float*) take(256 * 4);
  float*  b_ab2  = (float*) take(256 * 4);

  const int T = 256;
  dim3 blk(T);
  auto nb = [&](long n) { return dim3((unsigned)((n + T - 1) / T)); };
  const unsigned gemm_blocks = (unsigned)((N + 127) / 128);
  const float invN = 1.0f / (float)N;
  const long NH4 = NH / 4, GH4 = GH / 4;

  // ---- encode + degree ----
  k_zero4      <<<nb(N / 4), blk, 0, stream>>>((float4*)b_inv, N / 4);
  k_atom_encode<<<nb(NH4),   blk, 0, stream>>>(x, (const float4*)atom_emb, (float4*)b_h, NH4);
  k_deg        <<<nb(E),     blk, 0, stream>>>(erow, b_inv, E);
  k_inv_inplace<<<nb(N),     blk, 0, stream>>>(b_inv, N);

  // ---- layers ----
  for (int l = 0; l < NLAYERS; ++l) {
    k_zero4       <<<nb(NH4), blk, 0, stream>>>((float4*)b_acc, NH4);
    k_bond_scatter<<<nb(EH),  blk, 0, stream>>>(eattr, erow, bond_emb + (size_t)l * 3 * 8 * HD,
                                                b_acc, EH);
    k_hin         <<<nb(NH4), blk, 0, stream>>>((const float4*)b_h, (float4*)b_acc, b_inv,
                                                (float4*)b_hin, NH4);
    k_aggr_scatter<<<nb(EH),  blk, 0, stream>>>(erow, ecol, b_hin, b_acc, EH);

    // GEMM1: y1 = ((1+eps)*hin + acc) @ w1 ; BN1 stats fused; y1 -> b_h
    k_prep_w<<<dim3(64), blk, 0, stream>>>(w1 + (size_t)l * HD * HD, b_wt, b_st);
    k_gemm128<1, true, false><<<dim3(gemm_blocks), blk, 0, stream>>>(
        b_hin, b_acc, eps, l, b_wt, nullptr, b_h, b_st, N);
    k_bn_finalize<<<dim3(1), dim3(128), 0, stream>>>(b_st, bn1_g + (size_t)l * HD,
                                                     bn1_b + (size_t)l * HD, b_ab1, invN);

    // GEMM2: y2 = relu(bn1(y1)) @ w2 ; BN2 stats fused; y2 -> b_hin
    k_prep_w<<<dim3(64), blk, 0, stream>>>(w2 + (size_t)l * HD * HD, b_wt, b_st);
    k_gemm128<2, true, false><<<dim3(gemm_blocks), blk, 0, stream>>>(
        b_h, nullptr, b_ab1, 0, b_wt, nullptr, b_hin, b_st, N);
    k_bn_finalize<<<dim3(1), dim3(128), 0, stream>>>(b_st, bn_g + (size_t)l * HD,
                                                     bn_b + (size_t)l * HD, b_ab2, invN);

    // h = bn2(y2) (+relu except last layer) -> b_h
    k_bn_apply<<<nb(NH4), blk, 0, stream>>>((const float4*)b_hin, b_ab2, (float4*)b_h, NH4,
                                            (l < NLAYERS - 1) ? 1 : 0);
  }

  // ---- pooling + classifier ----
  k_zero4<<<nb(GH4), blk, 0, stream>>>((float4*)b_pool, GH4);
  k_pool <<<nb(NH),  blk, 0, stream>>>(batch, b_h, b_pool, NH);

  k_prep_w<<<dim3(64), blk, 0, stream>>>(cw1, b_wt, b_st);
  k_gemm128<0, false, true><<<dim3(Gn / 128), blk, 0, stream>>>(
      b_pool, nullptr, nullptr, 0, b_wt, cb1, b_hin, b_st, Gn);
  k_final_dot<<<dim3(Gn / 8), blk, 0, stream>>>(b_hin, cw2, cb2, out, Gn);
}
